// StoryModel_27075473834368
// MI455X (gfx1250) — matrix operand order repaired and checked by reading, not compiled, not run on hardware
//
#include <hip/hip_runtime.h>
#include <math.h>

// ---------------------------------------------------------------------------
// Model constants (must match reference)
// ---------------------------------------------------------------------------
#define C_V   32000
#define C_H   768
#define C_NH  12
#define C_HD  64
#define C_L   6
#define C_FF  3072
#define C_B   2
#define C_T   2048
#define C_M   (C_B * C_T)   // 4096 tokens

typedef __bf16 bf16_t;
typedef __attribute__((ext_vector_type(16))) __bf16 bf16x16;
typedef __attribute__((ext_vector_type(8)))  __bf16 bf16x8;
typedef __attribute__((ext_vector_type(8)))  float  f32x8;

// TDM descriptor SGPR-group vector types (per compile-probe signatures)
typedef __attribute__((ext_vector_type(4))) unsigned int tdm_u32x4;
typedef __attribute__((ext_vector_type(8))) int          tdm_i32x8;
typedef __attribute__((ext_vector_type(4))) int          tdm_i32x4;

// Tensor Data Mover availability (device pass only; host pass preprocesses out)
#if defined(__HIP_DEVICE_COMPILE__) && __has_builtin(__builtin_amdgcn_tensor_load_to_lds)
#define USE_TDM 1
#else
#define USE_TDM 0
#endif

#if USE_TDM
// Issue one 2-D TDM tile load: global (row-major, stride in g1) -> LDS.
// The trailing asm memory-clobber models the DMA's LDS writes, which the
// intrinsic itself does not expose to alias analysis.
static __device__ __forceinline__ void tdm_load(const void* g, unsigned lds,
                                                tdm_i32x8 g1) {
  const tdm_i32x4 gz = {0, 0, 0, 0};
  const unsigned long long ga = (unsigned long long)g;
  tdm_u32x4 g0;
  g0.x = 1u;                               // count=1 (valid descriptor)
  g0.y = lds;                              // lds_addr
  g0.z = (unsigned)ga;                     // global_addr[31:0]
  g0.w = (unsigned)(ga >> 32) | (2u << 30);// global_addr[56:32] | type=2
#if __clang_major__ >= 23
  const tdm_i32x8 gz8 = {0, 0, 0, 0, 0, 0, 0, 0};
  __builtin_amdgcn_tensor_load_to_lds(g0, g1, gz, gz, gz8, 0);
#else
  __builtin_amdgcn_tensor_load_to_lds(g0, g1, gz, gz, 0);
#endif
  asm volatile("" ::: "memory");
}
#endif

static __device__ __forceinline__ bf16x16 join8(bf16x8 lo, bf16x8 hi) {
  return __builtin_shufflevector(lo, hi, 0,1,2,3,4,5,6,7,8,9,10,11,12,13,14,15);
}

#define WMMA_BF16(a, b, c) \
  __builtin_amdgcn_wmma_f32_16x16x32_bf16(false, (a), false, (b), (short)0, (c), false, false)

// ---------------------------------------------------------------------------
// fp32 -> bf16 cast kernel (tok_emb: [V,H] stays as-is; it IS B^T for lm_head)
// ---------------------------------------------------------------------------
__global__ void cvt_f32_bf16(const float* __restrict__ s, bf16_t* __restrict__ d,
                             long long n) {
  long long i = (long long)blockIdx.x * blockDim.x + threadIdx.x;
  long long stride = (long long)gridDim.x * blockDim.x;
  for (; i < n; i += stride) d[i] = (bf16_t)s[i];
}

// Per-layer weight precast WITH transpose: s=[L][K][N] f32 -> d=[L][N][K] bf16.
// B^T tiles are then contiguous along K => direct TDM tile loads, no LDS
// transpose shuffling in the GEMM hot loop.
__global__ void transpose_w_f32_bf16(const float* __restrict__ s, bf16_t* __restrict__ d,
                                     int Kd, int Nd, int layers) {
  const long long per = (long long)Kd * Nd;
  const long long tot = per * layers;
  long long i = (long long)blockIdx.x * blockDim.x + threadIdx.x;
  long long stride = (long long)gridDim.x * blockDim.x;
  for (; i < tot; i += stride) {
    long long l = i / per;
    long long r = i % per;
    long long n = r / Kd;
    long long k = r % Kd;
    d[i] = (bf16_t)s[l * per + k * Nd + n];
  }
}

// ---------------------------------------------------------------------------
// x[b,t,:] = tok_emb[ids[b,t],:] + pos_emb[t,:]
// ---------------------------------------------------------------------------
__global__ __launch_bounds__(256)
void embed_kernel(const int* __restrict__ ids, const float* __restrict__ tok,
                  const float* __restrict__ pos, float* __restrict__ x) {
  const int bt = blockIdx.x;
  const int t  = bt % C_T;
  const int id = ids[bt];
  const float* tr = tok + (size_t)id * C_H;
  const float* pr = pos + (size_t)t * C_H;
  float* xr = x + (size_t)bt * C_H;
  for (int i = threadIdx.x; i < C_H; i += 256) xr[i] = tr[i] + pr[i];
}

// ---------------------------------------------------------------------------
// LayerNorm over last dim (768); writes bf16 activations for WMMA GEMMs
// ---------------------------------------------------------------------------
__global__ __launch_bounds__(256)
void layernorm_kernel(const float* __restrict__ x, const float* __restrict__ w,
                      const float* __restrict__ b, bf16_t* __restrict__ out) {
  const int row = blockIdx.x;
  const int tid = threadIdx.x;
  const float* xr = x + (size_t)row * C_H;
  __shared__ float red[256];

  float s = 0.f;
  for (int i = tid; i < C_H; i += 256) s += xr[i];
  red[tid] = s;
  __syncthreads();
  for (int st = 128; st > 0; st >>= 1) {
    if (tid < st) red[tid] += red[tid + st];
    __syncthreads();
  }
  const float mean = red[0] * (1.0f / C_H);
  __syncthreads();

  float v = 0.f;
  for (int i = tid; i < C_H; i += 256) {
    float d = xr[i] - mean;
    v += d * d;
  }
  red[tid] = v;
  __syncthreads();
  for (int st = 128; st > 0; st >>= 1) {
    if (tid < st) red[tid] += red[tid + st];
    __syncthreads();
  }
  const float rstd = rsqrtf(red[0] * (1.0f / C_H) + 1e-5f);

  bf16_t* orow = out + (size_t)row * C_H;
  for (int i = tid; i < C_H; i += 256)
    orow[i] = (bf16_t)((xr[i] - mean) * rstd * w[i] + b[i]);
}

// ---------------------------------------------------------------------------
// bf16 WMMA GEMM:  out = act(A[M,K] @ Bt[N,K]^T + bias) (+ residual)
//   block tile 128x64, TILE_K=32; 256 threads = 8 waves, each wave 32x32.
//   Double-buffered LDS; both tiles fetched by the Tensor Data Mover from
//   wave 0 with software pipelining (issue t+1, s_wait_tensorcnt 2 => tile t
//   resident, barrier, pure ds_load_b128 + v_wmma compute).
//   store_vt: scatter bf16 output as [B,NH,HD,T] (pre-transposed V for attn).
// ---------------------------------------------------------------------------
struct GemmSmem {
  __align__(16) bf16_t A[2][128 * 32];   // [buf][row][k]
  __align__(16) bf16_t BT[2][64 * 32];   // [buf][n][k]
};

__global__ __launch_bounds__(256)
void gemm_bf16(const bf16_t* __restrict__ A, const bf16_t* __restrict__ Bt,
               const float* __restrict__ bias, const float* residual,
               float* outF, bf16_t* outB,
               int M, int N, int K, int gelu, int store_vt) {
  __shared__ GemmSmem sm;

  const int tid  = threadIdx.x;
  const int lane = tid & 31;
  const int wave = tid >> 5;                 // 0..7
  const int wm   = (wave >> 1) * 32;         // 0,32,64,96
  const int wn   = (wave & 1) * 32;          // 0,32
  const int bm   = blockIdx.y * 128;
  const int bn   = blockIdx.x * 64;
  const int half = (lane < 16) ? 0 : 1;
  const int lr   = lane & 15;
  const int nk   = K >> 5;

  f32x8 acc[2][2] = {};

  const bf16_t* Atile = A + (size_t)bm * K;    // 128 x K, row stride K
  const bf16_t* Btile = Bt + (size_t)bn * K;   // 64 x K,  row stride K

#if USE_TDM
  // Mark the LDS buffers as written objects (variable-index volatile stores):
  // the DMA engine writes them behind the compiler's back, and without this
  // LLVM folds the fragment ds_loads to undef and deletes them.
  {
    volatile bf16_t* pa = (volatile bf16_t*)sm.A[0];
    volatile bf16_t* pb = (volatile bf16_t*)sm.BT[0];
    pa[tid] = (bf16_t)0.f;
    pb[tid] = (bf16_t)0.f;
  }
  __syncthreads();

  // D# group1 (invariant): data_size=2B, huge tensor dims (no OOB clipping),
  // tile_dim0=32 along K, dim0 stride = K elements.
  const tdm_i32x8 g1A = {0x10000, 0, 0x4000, 0x4000 | (32 << 16), 128, K, 0, 0};
  const tdm_i32x8 g1B = {0x10000, 0, 0x4000, 0x4000 | (32 << 16),  64, K, 0, 0};
  const unsigned ldsA = (unsigned)__builtin_offsetof(GemmSmem, A);
  const unsigned ldsB = (unsigned)__builtin_offsetof(GemmSmem, BT);

  if (wave == 0) {               // prologue: DMA tile 0 into buffer 0
    tdm_load(Atile, ldsA, g1A);
    tdm_load(Btile, ldsB, g1B);
  }
#endif

  for (int t = 0; t < nk; ++t) {
    const int k0 = t << 5;
    const int buf = t & 1;

#if USE_TDM
    if (wave == 0) {
      if (t + 1 < nk) {          // pipeline: issue tile t+1, wait for tile t
        const int nb = (t + 1) & 1;
        tdm_load(Atile + k0 + 32, ldsA + nb * (128 * 32 * 2), g1A);
        tdm_load(Btile + k0 + 32, ldsB + nb * (64 * 32 * 2), g1B);
        __builtin_amdgcn_s_wait_tensorcnt(2);
      } else {
        __builtin_amdgcn_s_wait_tensorcnt(0);
      }
    }
#else
    // Fallback synchronous staging (batched loads, then vector LDS stores)
    const int idx0 = tid, idx1 = tid + 256;
    const uint4 av0 = *(const uint4*)(Atile + (size_t)(idx0 >> 2) * K + k0 + (idx0 & 3) * 8);
    const uint4 av1 = *(const uint4*)(Atile + (size_t)(idx1 >> 2) * K + k0 + (idx1 & 3) * 8);
    const uint4 bv0 = *(const uint4*)(Btile + (size_t)(tid >> 2) * K + k0 + (tid & 3) * 8);
    *(uint4*)(sm.A[buf] + (idx0 >> 2) * 32 + (idx0 & 3) * 8) = av0;
    *(uint4*)(sm.A[buf] + (idx1 >> 2) * 32 + (idx1 & 3) * 8) = av1;
    *(uint4*)(sm.BT[buf] + (tid >> 2) * 32 + (tid & 3) * 8) = bv0;
#endif
    if (t + 2 < nk) {            // L2 warm-up ahead of the DMA engine
      __builtin_prefetch(Atile + (size_t)(tid >> 1) * K + k0 + 64, 0, 1);
      __builtin_prefetch(Btile + (size_t)(tid >> 2) * K + k0 + 64, 0, 1);
    }
    __syncthreads();             // tile t resident for all 8 waves
    asm volatile("" ::: "memory"); // LDS contents changed by the DMA engine

    bf16x16 afrag[2], bfrag[2];
#pragma unroll
    for (int mi = 0; mi < 2; ++mi) {
      const bf16_t* p = sm.A[buf] + (wm + mi * 16 + lr) * 32 + half * 8;
      afrag[mi] = join8(*(const bf16x8*)p, *(const bf16x8*)(p + 16));
    }
#pragma unroll
    for (int ni = 0; ni < 2; ++ni) {
      const bf16_t* p = sm.BT[buf] + (wn + ni * 16 + lr) * 32 + half * 8;
      bfrag[ni] = join8(*(const bf16x8*)p, *(const bf16x8*)(p + 16));
    }
#pragma unroll
    for (int mi = 0; mi < 2; ++mi)
#pragma unroll
      for (int ni = 0; ni < 2; ++ni)
        acc[mi][ni] = WMMA_BF16(afrag[mi], bfrag[ni], acc[mi][ni]);
    __syncthreads();             // all reads done; buffer may be overwritten
  }

  // ---- epilogue: bias (+gelu) (+residual), fp32 / bf16 / V-transposed stores
#pragma unroll
  for (int mi = 0; mi < 2; ++mi) {
#pragma unroll
    for (int ni = 0; ni < 2; ++ni) {
      const int n = bn + wn + ni * 16 + lr;
      const float bvv = bias ? bias[n] : 0.0f;
#pragma unroll
      for (int i = 0; i < 8; ++i) {
        const int m = bm + wm + mi * 16 + i + half * 8;
        float val = acc[mi][ni][i] + bvv;
        if (gelu) val = 0.5f * val * (1.0f + erff(val * 0.70710678118654752f));
        const size_t offo = (size_t)m * N + n;
        if (residual) val += residual[offo];
        if (outF) outF[offo] = val;
        if (outB) {
          if (store_vt) {        // [B,T,H] -> [B,NH,HD,T]
            const int bb = m >> 11, tt = m & (C_T - 1);
            const int hh = n >> 6, dd = n & (C_HD - 1);
            outB[(((size_t)bb * C_NH + hh) * C_HD + dd) * C_T + tt] = (bf16_t)val;
          } else {
            outB[offo] = (bf16_t)val;
          }
        }
      }
    }
  }
}

// ---------------------------------------------------------------------------
// Flash-attention: one wave per (b, head, 16-query block). Key blocks of 32.
// S = Q K^T via WMMA (K rows contiguous -> direct global B-fragments),
// online softmax through LDS, V pre-transposed [B,NH,HD,T] -> P.V B-fragments
// are contiguous global loads (no LDS staging for V).
// ---------------------------------------------------------------------------
__global__ __launch_bounds__(32)
void attention_kernel(const bf16_t* __restrict__ q, const bf16_t* __restrict__ k,
                      const bf16_t* __restrict__ vt, bf16_t* __restrict__ out) {
  const int bid = blockIdx.x;
  const int qi  = bid & 127;                  // T/16 = 128 query blocks
  const int h   = (bid >> 7) % C_NH;
  const int b   = bid / (C_NH * 128);
  const int qbase = qi * 16;

  const int lane = threadIdx.x;
  const int half = (lane < 16) ? 0 : 1;
  const int lr   = lane & 15;

  __shared__ __align__(16) float  sS[16 * 32];
  __shared__ __align__(16) bf16_t sP[16 * 32];
  __shared__ __align__(32) float  sAlpha[16];
  __shared__ __align__(32) float  sL[16];

  const bf16_t* qp  = q  + (size_t)b * C_T * C_H + (size_t)h * C_HD;
  const bf16_t* kp  = k  + (size_t)b * C_T * C_H + (size_t)h * C_HD;
  const bf16_t* vtp = vt + ((size_t)b * C_NH + h) * C_HD * C_T;  // [HD][T]

  // Q A-fragments for the two 32-wide k-steps over HD=64
  bf16x16 qf[2];
#pragma unroll
  for (int s = 0; s < 2; ++s) {
    const bf16_t* p = qp + (size_t)(qbase + lr) * C_H + s * 32 + half * 8;
    qf[s] = join8(*(const bf16x8*)p, *(const bf16x8*)(p + 16));
  }

  f32x8 oacc[4] = {};
  float mrow = -3.0e38f, lrow = 0.0f;         // valid on lanes 0..15 (row = lane)
  const float scale = 0.125f;                 // 1/sqrt(64)

  for (int jb = 0; jb <= qbase + 15; jb += 32) {
    // ---- S = Q K^T for 32 keys (two 16-wide n-tiles)
    f32x8 s0 = {}, s1 = {};
#pragma unroll
    for (int s = 0; s < 2; ++s) {
      const bf16_t* p0 = kp + (size_t)(jb + lr) * C_H + s * 32 + half * 8;
      bf16x16 kf0 = join8(*(const bf16x8*)p0, *(const bf16x8*)(p0 + 16));
      const bf16_t* p1 = kp + (size_t)(jb + 16 + lr) * C_H + s * 32 + half * 8;
      bf16x16 kf1 = join8(*(const bf16x8*)p1, *(const bf16x8*)(p1 + 16));
      s0 = WMMA_BF16(qf[s], kf0, s0);
      s1 = WMMA_BF16(qf[s], kf1, s1);
    }
    // scatter scaled scores to LDS (C-layout -> [row][col])
#pragma unroll
    for (int i = 0; i < 8; ++i) {
      sS[(i + half * 8) * 32 + lr]      = s0[i] * scale;
      sS[(i + half * 8) * 32 + 16 + lr] = s1[i] * scale;
    }
    __syncthreads();

    // ---- online softmax (lanes 0..15 own one query row each)
    if (lane < 16) {
      const int r = lane;
      float xs[32];
      float mx = mrow;
#pragma unroll
      for (int c = 0; c < 32; ++c) {
        float sv = sS[r * 32 + c];
        if (jb + c > qbase + r) sv = -3.0e38f;  // causal mask
        xs[c] = sv;
        mx = fmaxf(mx, sv);
      }
      const float alpha = __expf(mrow - mx);
      float sum = 0.f;
#pragma unroll
      for (int c = 0; c < 32; ++c) {
        float p = __expf(xs[c] - mx);           // masked -> underflows to 0
        sP[r * 32 + c] = (bf16_t)p;
        sum += p;
      }
      lrow = lrow * alpha + sum;
      mrow = mx;
      sAlpha[r] = alpha;
    }
    __syncthreads();

    // ---- rescale running O by per-row alpha
    {
      f32x8 av = *(const f32x8*)(sAlpha + half * 8);
#pragma unroll
      for (int d = 0; d < 4; ++d)
#pragma unroll
        for (int i = 0; i < 8; ++i) oacc[d][i] *= av[i];
    }
    // ---- P A-fragment (K = 32 keys)
    bf16x16 pf;
    {
      const bf16_t* pp = sP + lr * 32 + half * 8;
      pf = join8(*(const bf16x8*)pp, *(const bf16x8*)(pp + 16));
    }
    // ---- O += P V : B-fragments straight from pre-transposed V rows
#pragma unroll
    for (int d = 0; d < 4; ++d) {
      const bf16_t* pvt = vtp + (size_t)(d * 16 + lr) * C_T + jb + half * 8;
      bf16x16 vf = join8(*(const bf16x8*)pvt, *(const bf16x8*)(pvt + 16));
      oacc[d] = WMMA_BF16(pf, vf, oacc[d]);
    }
    __syncthreads();
  }

  // ---- normalize by l and store bf16 [B,T,H]
  if (lane < 16) sL[lane] = lrow;
  __syncthreads();
  f32x8 lv = *(const f32x8*)(sL + half * 8);
#pragma unroll
  for (int i = 0; i < 8; ++i) {
    const float inv = 1.0f / lv[i];
    const int row = qbase + i + half * 8;
    bf16_t* orow = out + ((size_t)b * C_T + row) * C_H + (size_t)h * C_HD;
#pragma unroll
    for (int d = 0; d < 4; ++d) orow[d * 16 + lr] = (bf16_t)(oacc[d][i] * inv);
  }
}

// ---------------------------------------------------------------------------
// Host orchestration
// ---------------------------------------------------------------------------
extern "C" void kernel_launch(void* const* d_in, const int* in_sizes, int n_in,
                              void* d_out, int out_size, void* d_ws, size_t ws_size,
                              hipStream_t stream) {
  const int*   ids  = (const int*)  d_in[0];
  const float* tok  = (const float*)d_in[1];
  const float* pos  = (const float*)d_in[2];
  const float* Wq   = (const float*)d_in[3];
  const float* bq   = (const float*)d_in[4];
  const float* Wk   = (const float*)d_in[5];
  const float* bk   = (const float*)d_in[6];
  const float* Wv   = (const float*)d_in[7];
  const float* bv   = (const float*)d_in[8];
  const float* Wo   = (const float*)d_in[9];
  const float* bo   = (const float*)d_in[10];
  const float* ln1w = (const float*)d_in[11];
  const float* ln1b = (const float*)d_in[12];
  const float* ln2w = (const float*)d_in[13];
  const float* ln2b = (const float*)d_in[14];
  const float* W1   = (const float*)d_in[15];
  const float* b1   = (const float*)d_in[16];
  const float* W2   = (const float*)d_in[17];
  const float* b2   = (const float*)d_in[18];
  const float* lnfw = (const float*)d_in[19];
  const float* lnfb = (const float*)d_in[20];

  char* ws = (char*)d_ws;
  size_t off = 0;
  auto carve = [&](size_t bytes) -> char* {
    char* p = ws + off;
    off = (off + bytes + 255) & ~(size_t)255;
    return p;
  };

  float*  x    = (float*) carve((size_t)C_M * C_H * 4);   // residual stream (fp32)
  bf16_t* hln  = (bf16_t*)carve((size_t)C_M * C_H * 2);   // LN output (bf16)
  bf16_t* qb   = (bf16_t*)carve((size_t)C_M * C_H * 2);
  bf16_t* kbb  = (bf16_t*)carve((size_t)C_M * C_H * 2);
  bf16_t* vtb  = (bf16_t*)carve((size_t)C_M * C_H * 2);   // V as [B,NH,HD,T]
  bf16_t* attn = (bf16_t*)carve((size_t)C_M * C_H * 2);
  bf16_t* ffb  = (bf16_t*)carve((size_t)C_M * C_FF * 2);
  bf16_t* wqB  = (bf16_t*)carve((size_t)C_L * C_H * C_H * 2);   // [L][N][K] bf16
  bf16_t* wkB  = (bf16_t*)carve((size_t)C_L * C_H * C_H * 2);
  bf16_t* wvB  = (bf16_t*)carve((size_t)C_L * C_H * C_H * 2);
  bf16_t* woB  = (bf16_t*)carve((size_t)C_L * C_H * C_H * 2);
  bf16_t* w1B  = (bf16_t*)carve((size_t)C_L * C_H * C_FF * 2); // [L][FF][H]
  bf16_t* w2B  = (bf16_t*)carve((size_t)C_L * C_FF * C_H * 2); // [L][H][FF]
  bf16_t* tokB = (bf16_t*)carve((size_t)C_V * C_H * 2);        // [V][H] = lm_head B^T

  // --- precast weights to bf16, transposed to [N][K] (deterministic each call)
  transpose_w_f32_bf16<<<2048, 256, 0, stream>>>(Wq, wqB, C_H, C_H, C_L);
  transpose_w_f32_bf16<<<2048, 256, 0, stream>>>(Wk, wkB, C_H, C_H, C_L);
  transpose_w_f32_bf16<<<2048, 256, 0, stream>>>(Wv, wvB, C_H, C_H, C_L);
  transpose_w_f32_bf16<<<2048, 256, 0, stream>>>(Wo, woB, C_H, C_H, C_L);
  transpose_w_f32_bf16<<<2048, 256, 0, stream>>>(W1, w1B, C_H, C_FF, C_L);
  transpose_w_f32_bf16<<<2048, 256, 0, stream>>>(W2, w2B, C_FF, C_H, C_L);
  cvt_f32_bf16<<<2048, 256, 0, stream>>>(tok, tokB, (long long)C_V * C_H);

  // --- embeddings
  embed_kernel<<<C_M, 256, 0, stream>>>(ids, tok, pos, x);

  const dim3 gH (C_H  / 64, C_M / 128);   // (12, 32)
  const dim3 gFF(C_FF / 64, C_M / 128);   // (48, 32)
  const dim3 gV (C_V  / 64, C_M / 128);   // (500, 32)

  for (int l = 0; l < C_L; ++l) {
    const size_t oHH = (size_t)l * C_H * C_H;
    const size_t oHF = (size_t)l * C_H * C_FF;
    const size_t oH  = (size_t)l * C_H;
    const size_t oF  = (size_t)l * C_FF;

    layernorm_kernel<<<C_M, 256, 0, stream>>>(x, ln1w + oH, ln1b + oH, hln);
    gemm_bf16<<<gH, 256, 0, stream>>>(hln, wqB + oHH, bq + oH, nullptr, nullptr, qb,
                                      C_M, C_H, C_H, 0, 0);
    gemm_bf16<<<gH, 256, 0, stream>>>(hln, wkB + oHH, bk + oH, nullptr, nullptr, kbb,
                                      C_M, C_H, C_H, 0, 0);
    gemm_bf16<<<gH, 256, 0, stream>>>(hln, wvB + oHH, bv + oH, nullptr, nullptr, vtb,
                                      C_M, C_H, C_H, 0, 1 /*store V transposed*/);
    attention_kernel<<<C_B * C_NH * (C_T / 16), 32, 0, stream>>>(qb, kbb, vtb, attn);
    gemm_bf16<<<gH, 256, 0, stream>>>(attn, woB + oHH, bo + oH, x, x, nullptr,
                                      C_M, C_H, C_H, 0, 0);
    layernorm_kernel<<<C_M, 256, 0, stream>>>(x, ln2w + oH, ln2b + oH, hln);
    gemm_bf16<<<gFF, 256, 0, stream>>>(hln, w1B + oHF, b1 + oF, nullptr, nullptr, ffb,
                                       C_M, C_FF, C_H, 1 /*gelu*/, 0);
    gemm_bf16<<<gH, 256, 0, stream>>>(ffb, w2B + oHF, b2 + oH, x, x, nullptr,
                                      C_M, C_H, C_FF, 0, 0);
  }

  layernorm_kernel<<<C_M, 256, 0, stream>>>(x, lnfw, lnfb, hln);
  gemm_bf16<<<gV, 256, 0, stream>>>(hln, tokB, nullptr, nullptr, (float*)d_out, nullptr,
                                    C_M, C_V, C_H, 0, 0);
}
